// WaveNet_77300821393880
// MI455X (gfx1250) — matrix operand order repaired and checked by reading, not compiled, not run on hardware
//
#include <hip/hip_runtime.h>
#include <hip/hip_bf16.h>

// ---------------------------------------------------------------------------
// WaveNet forward for gfx1250 (MI455X). Compute-heavy residual blocks use
// v_wmma_f32_16x16x32_f16 (wave32 WMMA); memory-bound dense layer streams f32.
// ---------------------------------------------------------------------------

typedef __attribute__((ext_vector_type(16))) _Float16 v16h;
typedef __attribute__((ext_vector_type(8)))  _Float16 v8h;
typedef __attribute__((ext_vector_type(8)))  float    v8f;

#define SK   2050
#define CCH  256

__device__ __forceinline__ v8f wmma_f16(v16h a, v16h b, v8f c) {
  // D = A(16x32) * B(32x16) + C, f32 accumulate
  return __builtin_amdgcn_wmma_f32_16x16x32_f16(
      /*neg_a=*/false, a, /*neg_b=*/false, b,
      /*c_mod=*/(short)0, c, /*reuse_a=*/false, /*reuse_b=*/false);
}

// A-tile per-lane load (16x32 f16, row-major source with K contiguous):
// lane<16: M=lane, K = {k0..k0+15} with k0 = kk
// lane>=16: M=lane-16, k0 = kk+16  (caller bakes +half*16 into the pointer)
__device__ __forceinline__ v16h ld_a16(const _Float16* p) {
  v8h lo = *(const v8h*)(p);
  v8h hi = *(const v8h*)(p + 8);
  v16h r;
#pragma unroll
  for (int j = 0; j < 8; ++j) { r[j] = lo[j]; r[j + 8] = hi[j]; }
  return r;
}

// ---------------------------------------------------------------------------
// Weight convert + transpose: f32 [nmat][256(Cin)][256(Cout)] -> f16 [nmat][Cout][Cin]
// ---------------------------------------------------------------------------
__global__ void k_w2h_t(const float* __restrict__ src, _Float16* __restrict__ dst,
                        int nmat, long srcStride, long dstStride) {
  long i = (long)blockIdx.x * 256 + threadIdx.x;   // over nmat*65536
  long m = i >> 16;
  int rem = (int)(i & 65535);
  int ci = rem >> 8;
  int co = rem & 255;
  if (m < nmat)
    dst[m * dstStride + (long)co * 256 + ci] =
        (_Float16)src[m * srcStride + (long)ci * 256 + co];
}

__global__ void k_zero(float* __restrict__ p, long n) {
  long i = (long)blockIdx.x * 256 + threadIdx.x;
  if (i < n) p[i] = 0.0f;
}

// initial 1x1 causal conv, Cin = 1: x[b][t][c] = in[b][t]*cw[c] + cb[c]  (f16 out)
__global__ void k_init(const float* __restrict__ in, const float* __restrict__ cw,
                       const float* __restrict__ cb, _Float16* __restrict__ x) {
  long i = (long)blockIdx.x * 256 + threadIdx.x;   // over 8*4096*256
  long bt = i >> 8;
  int c = (int)(i & 255);
  x[i] = (_Float16)(in[bt] * cw[c] + cb[c]);
}

// ---------------------------------------------------------------------------
// One residual block, fully fused:
//   h = x[t] @ W0 + x[t+d] @ W1 + bdil            (WMMA phase 1)
//   g = tanh(h)*sigmoid(h)          -> LDS (f16)
//   res = g @ Wr + bres ; skip = g @ Ws + bskip   (WMMA phase 2, A from LDS)
//   xout = res + g (f16) ; skip_acc[last SK] += skip (f32)
// Grid: (ceil(Tout/16), 8 batches) x 128 threads (4 waves, 4 n-tiles each)
// ---------------------------------------------------------------------------
__global__ __launch_bounds__(128) void k_block(
    const _Float16* __restrict__ xin, _Float16* __restrict__ xout,
    const _Float16* __restrict__ w0t, const _Float16* __restrict__ w1t,
    const _Float16* __restrict__ wrt, const _Float16* __restrict__ wst,
    const float* __restrict__ bdil, const float* __restrict__ bres,
    const float* __restrict__ bskip, float* __restrict__ skip_acc,
    int Tin, int Tout, int d, int skip_start) {
  __shared__ _Float16 g[16][272];   // 16 rows x 256 ch, padded stride (544B rows)

  const int bb   = blockIdx.y;
  const int m0   = blockIdx.x * 16;
  const int lane = threadIdx.x & 31;
  const int wv   = threadIdx.x >> 5;
  const int hf   = lane >> 4;       // 0 / 1 half-wave
  const int lrow = lane & 15;

  const _Float16* xb = xin + (size_t)bb * Tin * CCH;
  int ra = m0 + lrow;       if (ra > Tin - 1) ra = Tin - 1;
  int rb = m0 + lrow + d;   if (rb > Tin - 1) rb = Tin - 1;

  // ---- phase 1: dilated conv (two GEMMs summed) ----
  v8f acc[4];
#pragma unroll
  for (int nt = 0; nt < 4; ++nt) acc[nt] = (v8f){};

  for (int kk = 0; kk < CCH; kk += 32) {
    v16h a0 = ld_a16(xb + (size_t)ra * CCH + kk + hf * 16);
    v16h a1 = ld_a16(xb + (size_t)rb * CCH + kk + hf * 16);
#pragma unroll
    for (int nt = 0; nt < 4; ++nt) {
      int n0 = (wv * 4 + nt) * 16;
      v16h b0 = *(const v16h*)(w0t + (size_t)(n0 + lrow) * CCH + kk + hf * 16);
      v16h b1 = *(const v16h*)(w1t + (size_t)(n0 + lrow) * CCH + kk + hf * 16);
      acc[nt] = wmma_f16(a0, b0, acc[nt]);
      acc[nt] = wmma_f16(a1, b1, acc[nt]);
    }
  }

  // gated activation -> LDS
#pragma unroll
  for (int nt = 0; nt < 4; ++nt) {
    int n = (wv * 4 + nt) * 16 + lrow;
    float bia = bdil[n];
#pragma unroll
    for (int r = 0; r < 8; ++r) {
      float h = acc[nt][r] + bia;
      float gg = tanhf(h) * (1.0f / (1.0f + expf(-h)));
      g[r + hf * 8][n] = (_Float16)gg;
    }
  }
  __syncthreads();

  // ---- phase 2: res + skip GEMMs, A from LDS ----
  v8f accR[4], accS[4];
#pragma unroll
  for (int nt = 0; nt < 4; ++nt) { accR[nt] = (v8f){}; accS[nt] = (v8f){}; }

  for (int kk = 0; kk < CCH; kk += 32) {
    v16h aL = ld_a16(&g[lrow][kk + hf * 16]);
#pragma unroll
    for (int nt = 0; nt < 4; ++nt) {
      int n0 = (wv * 4 + nt) * 16;
      v16h bR = *(const v16h*)(wrt + (size_t)(n0 + lrow) * CCH + kk + hf * 16);
      v16h bS = *(const v16h*)(wst + (size_t)(n0 + lrow) * CCH + kk + hf * 16);
      accR[nt] = wmma_f16(aL, bR, accR[nt]);
      accS[nt] = wmma_f16(aL, bS, accS[nt]);
    }
  }

  // ---- epilogue ----
#pragma unroll
  for (int nt = 0; nt < 4; ++nt) {
    int n = (wv * 4 + nt) * 16 + lrow;
    float br = bres[n], bs = bskip[n];
#pragma unroll
    for (int r = 0; r < 8; ++r) {
      int m = r + hf * 8;
      int t = m0 + m;
      if (t < Tout) {
        float gg = (float)g[m][n];
        xout[((size_t)bb * Tout + t) * CCH + n] = (_Float16)(accR[nt][r] + br + gg);
        int ts = t - skip_start;
        if (ts >= 0) {
          float* p = skip_acc + ((size_t)bb * SK + ts) * CCH + n;
          *p += accS[nt][r] + bs;   // tiles disjoint per workgroup; blocks sequential
        }
      }
    }
  }
}

// skip mean over 20 blocks + relu -> f16
__global__ void k_meanrelu(const float* __restrict__ sa, _Float16* __restrict__ s1, long n) {
  long i = (long)blockIdx.x * 256 + threadIdx.x;
  if (i < n) {
    float v = sa[i] * (1.0f / 20.0f);
    s1[i] = (_Float16)fmaxf(v, 0.0f);
  }
}

// generic [M,256] x [256,256] WMMA GEMM + bias (+optional relu), f16 in/out
__global__ __launch_bounds__(128) void k_gemm256(
    const _Float16* __restrict__ A, const _Float16* __restrict__ Wt,
    const float* __restrict__ bias, _Float16* __restrict__ out,
    int M, int dorelu) {
  const int m0   = blockIdx.x * 16;
  const int lane = threadIdx.x & 31;
  const int wv   = threadIdx.x >> 5;
  const int hf   = lane >> 4;
  const int lrow = lane & 15;

  int ra = m0 + lrow; if (ra > M - 1) ra = M - 1;

  v8f acc[4];
#pragma unroll
  for (int nt = 0; nt < 4; ++nt) acc[nt] = (v8f){};

  for (int kk = 0; kk < CCH; kk += 32) {
    v16h a = ld_a16(A + (size_t)ra * CCH + kk + hf * 16);
#pragma unroll
    for (int nt = 0; nt < 4; ++nt) {
      int n0 = (wv * 4 + nt) * 16;
      v16h bm = *(const v16h*)(Wt + (size_t)(n0 + lrow) * CCH + kk + hf * 16);
      acc[nt] = wmma_f16(a, bm, acc[nt]);
    }
  }

#pragma unroll
  for (int nt = 0; nt < 4; ++nt) {
    int n = (wv * 4 + nt) * 16 + lrow;
    float bia = bias[n];
#pragma unroll
    for (int r = 0; r < 8; ++r) {
      int t = m0 + r + hf * 8;
      if (t < M) {
        float v = acc[nt][r] + bia;
        if (dorelu) v = fmaxf(v, 0.0f);
        out[(size_t)t * CCH + n] = (_Float16)v;
      }
    }
  }
}

// dense: logits[b][nc] += sum_{k in chunk} f[b][k] * w[k][nc]  (BW-bound, f32 stream)
__global__ void k_dense(const _Float16* __restrict__ f, const float* __restrict__ w,
                        float* __restrict__ logits, long K, long chunk) {
  const int bb = blockIdx.y;
  const int nc = threadIdx.x;
  long k0 = (long)blockIdx.x * chunk;
  long k1 = k0 + chunk; if (k1 > K) k1 = K;
  const _Float16* fb = f + (size_t)bb * K;
  float acc = 0.0f;
  for (long k = k0; k < k1; ++k)
    acc += (float)fb[k] * w[(size_t)k * 256 + nc];   // coalesced across nc
  atomicAdd(&logits[bb * 256 + nc], acc);
}

__global__ void k_softmax(const float* __restrict__ logits, const float* __restrict__ bias,
                          float* __restrict__ out) {
  __shared__ float red[256];
  int bb = blockIdx.x, t = threadIdx.x;
  float v = logits[bb * 256 + t] + bias[t];
  red[t] = v; __syncthreads();
  for (int s = 128; s > 0; s >>= 1) { if (t < s) red[t] = fmaxf(red[t], red[t + s]); __syncthreads(); }
  float mx = red[0]; __syncthreads();
  float e = expf(v - mx);
  red[t] = e; __syncthreads();
  for (int s = 128; s > 0; s >>= 1) { if (t < s) red[t] += red[t + s]; __syncthreads(); }
  out[bb * 256 + t] = e / red[0];
}

// ---------------------------------------------------------------------------
extern "C" void kernel_launch(void* const* d_in, const int* in_sizes, int n_in,
                              void* d_out, int out_size, void* d_ws, size_t ws_size,
                              hipStream_t stream) {
  (void)in_sizes; (void)n_in; (void)out_size; (void)ws_size;

  const float* input_x  = (const float*)d_in[0];
  const float* causal_w = (const float*)d_in[1];
  const float* causal_b = (const float*)d_in[2];
  const float* dil_w    = (const float*)d_in[3];
  const float* dil_b    = (const float*)d_in[4];
  const float* res_w    = (const float*)d_in[5];
  const float* res_b    = (const float*)d_in[6];
  const float* skip_w   = (const float*)d_in[7];
  const float* skip_b   = (const float*)d_in[8];
  const float* cls_w    = (const float*)d_in[9];
  const float* cls_b    = (const float*)d_in[10];
  const float* dense_w  = (const float*)d_in[11];
  const float* dense_b  = (const float*)d_in[12];
  float* out = (float*)d_out;

  // workspace layout
  char* ws = (char*)d_ws;
  _Float16* X0   = (_Float16*)(ws);                       // 8*4096*256 f16
  _Float16* X1   = (_Float16*)(ws + 16777216);
  _Float16* W0T  = (_Float16*)(ws + 33554432);            // 20*256*256 f16 each
  _Float16* W1T  = (_Float16*)(ws + 36175872);
  _Float16* WRT  = (_Float16*)(ws + 38797312);
  _Float16* WST  = (_Float16*)(ws + 41418752);
  _Float16* WCT  = (_Float16*)(ws + 44040192);            // 256*256 f16
  float*    SACC = (float*)   (ws + 44171264);            // 8*2050*256 f32
  _Float16* S1   = (_Float16*)(ws + 60964864);            // 8*2050*256 f16
  _Float16* S2   = (_Float16*)(ws + 69361664);
  _Float16* S3   = (_Float16*)(ws + 77758464);
  float*    LOG  = (float*)   (ws + 86155264);            // 8*256 f32

  // 1) weight convert + transpose to f16 [Cout][Cin]
  k_w2h_t<<<20 * 256, 256, 0, stream>>>(dil_w,         W0T, 20, 131072, 65536);
  k_w2h_t<<<20 * 256, 256, 0, stream>>>(dil_w + 65536, W1T, 20, 131072, 65536);
  k_w2h_t<<<20 * 256, 256, 0, stream>>>(res_w,         WRT, 20, 65536,  65536);
  k_w2h_t<<<20 * 256, 256, 0, stream>>>(skip_w,        WST, 20, 65536,  65536);
  k_w2h_t<<<256,      256, 0, stream>>>(cls_w,         WCT, 1,  65536,  65536);

  // 2) zero accumulators (deterministic per call)
  k_zero<<<16400, 256, 0, stream>>>(SACC, (long)8 * SK * CCH);
  k_zero<<<8,     256, 0, stream>>>(LOG,  8 * 256);

  // 3) initial 1x1 causal conv (Cin=1) -> f16 activations
  k_init<<<8 * 4096, 256, 0, stream>>>(input_x, causal_w, causal_b, X0);

  // 4) 20 fused residual blocks (sequential dependency)
  static const int dils[20] = {1, 2, 4, 8, 16, 32, 64, 128, 256, 512,
                               1, 2, 4, 8, 16, 32, 64, 128, 256, 512};
  _Float16* xin = X0; _Float16* xout = X1;
  int Tin = 4096;
  for (int i = 0; i < 20; ++i) {
    int d = dils[i];
    int Tout = Tin - d;
    int skip_start = Tout - SK;
    dim3 grid((Tout + 15) / 16, 8);
    k_block<<<grid, 128, 0, stream>>>(
        xin, xout,
        W0T + (size_t)i * 65536, W1T + (size_t)i * 65536,
        WRT + (size_t)i * 65536, WST + (size_t)i * 65536,
        dil_b + i * 256, res_b + i * 256, skip_b + i * 256,
        SACC, Tin, Tout, d, skip_start);
    _Float16* tmp = xin; xin = xout; xout = tmp;
    Tin = Tout;
  }

  // 5) head: mean over blocks + relu, then shared 1x1 conv twice
  k_meanrelu<<<16400, 256, 0, stream>>>(SACC, S1, (long)8 * SK * CCH);
  k_gemm256<<<1025, 128, 0, stream>>>(S1, WCT, cls_b, S2, 8 * SK, /*relu=*/1);
  k_gemm256<<<1025, 128, 0, stream>>>(S2, WCT, cls_b, S3, 8 * SK, /*relu=*/0);

  // 6) dense (memory-bound: 537MB f32 weights streamed once) + softmax
  {
    const long K = (long)SK * CCH;     // 524800
    const long chunk = 1312;           // 400 chunks * 1312 = K
    dim3 grid(400, 8);
    k_dense<<<grid, 256, 0, stream>>>(S3, dense_w, LOG, K, chunk);
  }
  k_softmax<<<8, 256, 0, stream>>>(LOG, dense_b, out);
}
